// Multi_Attn_919123001389
// MI455X (gfx1250) — matrix operand order repaired and checked by reading, not compiled, not run on hardware
//
#include <hip/hip_runtime.h>

// Problem constants (from reference)
#define DIN   1024
#define SEQL  2048
#define NH    16
#define HD    64
#define NB    2

typedef _Float16 v8h  __attribute__((ext_vector_type(8)));
typedef _Float16 v16h __attribute__((ext_vector_type(16)));
typedef float    v8f  __attribute__((ext_vector_type(8)));
typedef float    v4f  __attribute__((ext_vector_type(4)));

static __device__ __forceinline__ v16h pack16(v8h lo, v8h hi) {
  v16h r;
#pragma unroll
  for (int i = 0; i < 8; ++i) { r[i] = lo[i]; r[i + 8] = hi[i]; }
  return r;
}

// ---------------------------------------------------------------------------
// Kernel 1: fused QKV projection.  grid (NH, (NB*SEQL)/64, 3), block 128.
//   z=0: Q = (query @ Wq) * (1/sqrt(64))  -> Qh[b][h][l][d]   (f16)
//   z=1: K =  key   @ Wk                  -> Kh[b][h][l][d]   (f16)
//   z=2: V = (value @ Wv) * (1/SEQL)      -> Vt[b][h][d][l]   (f16, transposed)
// 64x64 tile per block; double-buffered LDS stage -> one barrier per k-step.
// ---------------------------------------------------------------------------
__global__ __launch_bounds__(128) void proj_kernel(
    const float* __restrict__ Xq, const float* __restrict__ Xk, const float* __restrict__ Xv,
    const float* __restrict__ Wq, const float* __restrict__ Wk, const float* __restrict__ Wv,
    _Float16* __restrict__ Qh, _Float16* __restrict__ Kh, _Float16* __restrict__ Vt)
{
  __shared__ _Float16 lds_a[2][64 * 40];  // X tile: 64 rows x 32 k (stride 40 halves)
  __shared__ _Float16 lds_b[2][64 * 40];  // W tile transposed: 64 n x 32 k
  __shared__ _Float16 lds_c[64 * 72];     // output transpose staging

  const int t    = threadIdx.x;
  const int wid  = t >> 5;
  const int lane = t & 31;
  const int r16  = lane & 15;
  const int hi   = lane >> 4;       // 0: lanes 0-15, 1: lanes 16-31
  const int z    = blockIdx.z;
  const int h    = blockIdx.x;      // head == 64-wide N tile
  const int m0   = blockIdx.y * 64; // row tile within (NB*SEQL)

  const float* X = (z == 0) ? Xq : (z == 1) ? Xk : Xv;
  const float* W = (z == 0) ? Wq : (z == 1) ? Wk : Wv;

  auto stage = [&](int kt, int buf) {
    const float* Xt = X + (size_t)m0 * DIN + kt * 32;
    const float* Wt = W + (size_t)(kt * 32) * (NH * HD) + h * 64;
#pragma unroll
    for (int i = 0; i < 4; ++i) {
      int idx = t + i * 128;                 // float4 index 0..511
      // A tile: 64x32 f32 -> f16
      int row = idx >> 3, c4 = (idx & 7) * 4;
      v4f x = *(const v4f*)(Xt + (size_t)row * DIN + c4);
      _Float16* da = &lds_a[buf][row * 40 + c4];
      da[0] = (_Float16)x.x; da[1] = (_Float16)x.y;
      da[2] = (_Float16)x.z; da[3] = (_Float16)x.w;
      // B tile: 32x64 f32, store transposed (n-major) for contiguous fragments
      int k = idx >> 4, n4 = (idx & 15) * 4;
      v4f w = *(const v4f*)(Wt + (size_t)k * (NH * HD) + n4);
      lds_b[buf][(n4 + 0) * 40 + k] = (_Float16)w.x;
      lds_b[buf][(n4 + 1) * 40 + k] = (_Float16)w.y;
      lds_b[buf][(n4 + 2) * 40 + k] = (_Float16)w.z;
      lds_b[buf][(n4 + 3) * 40 + k] = (_Float16)w.w;
    }
  };

  v8f acc[4] = {};
  const int ra  = wid * 16 + r16;
  const int kbA = hi * 8;
  const int kbB = hi * 8;

  stage(0, 0);
  __syncthreads();
  for (int kt = 0; kt < 32; ++kt) {
    const int cur = kt & 1;
    if (kt + 1 < 32) stage(kt + 1, cur ^ 1);   // overlap next stage with WMMA
    v16h a = pack16(*(v8h*)&lds_a[cur][ra * 40 + kbA],
                    *(v8h*)&lds_a[cur][ra * 40 + kbA + 16]);
#pragma unroll
    for (int nt = 0; nt < 4; ++nt) {
      int n = nt * 16 + r16;
      v16h bb = pack16(*(v8h*)&lds_b[cur][n * 40 + kbB],
                       *(v8h*)&lds_b[cur][n * 40 + kbB + 16]);
      acc[nt] = __builtin_amdgcn_wmma_f32_16x16x32_f16(
          false, a, false, bb, (short)0, acc[nt], false, false);
    }
    __syncthreads();                            // one barrier per k-step
  }

  const float scale = (z == 0) ? 0.125f : (z == 2) ? (1.0f / (float)SEQL) : 1.0f;
  const int  b   = m0 >> 11;               // m0 / SEQL
  const int  l0  = m0 & (SEQL - 1);
  const bool isV = (z == 2);
  const size_t bh = (size_t)b * NH + h;

  // Scatter the C tile into LDS (f16) with a selected index (no branches):
  // Q/K: lds_c[row][d] (row-major);  V: lds_c[d][row] (transposed for Vt).
#pragma unroll
  for (int nt = 0; nt < 4; ++nt)
#pragma unroll
    for (int j = 0; j < 8; ++j) {
      int row = wid * 16 + j + hi * 8;     // C layout: VGPR j, lane-half selects +8
      int d   = nt * 16 + r16;
      int li  = isV ? (d * 72 + row) : (row * 72 + d);
      lds_c[li] = (_Float16)(acc[nt][j] * scale);
    }
  __syncthreads();
  _Float16* outQ = (z == 0) ? Qh : Kh;
#pragma unroll
  for (int i = 0; i < 4; ++i) {
    int idx = t + i * 128;                 // v8h units: 64 major x 8 chunks
    int a8 = idx >> 3, c8 = (idx & 7) * 8;
    v8h v = *(v8h*)&lds_c[a8 * 72 + c8];
    _Float16* gq = &outQ[(bh * SEQL + (size_t)(l0 + a8)) * HD + c8];
    _Float16* gv = &Vt[(bh * HD + a8) * SEQL + (size_t)(l0 + c8)];
    *(v8h*)(isV ? gv : gq) = v;
  }
}

// ---------------------------------------------------------------------------
// Kernel 2: attention. grid (SEQL/16, NH, NB), block 128 (4 waves).
// Block owns 16 query rows of one (b,h); wave w owns keys [512w, 512w+512).
// Pass 1  : S = Qs @ K^T (f16 WMMA) -> LDS (f16) + row max (fmax only).
// Pass 1.5: e = exp(s - M) -> back into LDS (exactly 1 exp per element) + row sums.
// Pass 2  : p = e * invS -> attn (non-temporal) and A-frags for O = P @ Vt.
// Dynamic LDS: 16 rows x 2064 halves = 66048 B; the 32B/row pad carries the
// per-wave row max (+0) and row sum (+4), 8B per wave.
// ---------------------------------------------------------------------------
__global__ __launch_bounds__(128) void attn_kernel(
    const _Float16* __restrict__ Qh, const _Float16* __restrict__ Kh,
    const _Float16* __restrict__ Vt,
    float* __restrict__ attn_v, float* __restrict__ attn)
{
  extern __shared__ char smem[];
  _Float16* Sbuf = (_Float16*)smem;
  const int SROW = 2064;                       // halves; 4128 B rows (16B aligned)

  const int t = threadIdx.x, wid = t >> 5, lane = t & 31;
  const int r = lane & 15, hi = lane >> 4;
  const int qt = blockIdx.x, h = blockIdx.y, b = blockIdx.z;
  const size_t bh = (size_t)b * NH + h;

  const _Float16* Qrow  = Qh + (bh * SEQL + (size_t)qt * 16) * HD;
  const _Float16* Kbase = Kh + bh * SEQL * (size_t)HD;
  const _Float16* Vbase = Vt + bh * (size_t)HD * SEQL;

  // Q A-fragments (16x32 each, contraction d: [0,32) and [32,64))
  v16h a0, a1;
  {
    const _Float16* q = Qrow + r * HD + hi * 8;
    a0 = pack16(*(v8h*)(q),      *(v8h*)(q + 16));
    a1 = pack16(*(v8h*)(q + 32), *(v8h*)(q + 48));
  }

  float m[8];
#pragma unroll
  for (int j = 0; j < 8; ++j) m[j] = -1e30f;

  const int c0 = wid * 512;
  // ---- Pass 1: scores into LDS + running row max (no exp) ----
  for (int kt = 0; kt < 32; ++kt) {
    const int key0 = c0 + kt * 16;
    const _Float16* kr = Kbase + (size_t)(key0 + r) * HD + hi * 8;
    __builtin_prefetch((const void*)(kr + 16 * HD), 0, 0);  // next key tile
    v16h b0 = pack16(*(v8h*)kr,        *(v8h*)(kr + 16));
    v16h b1 = pack16(*(v8h*)(kr + 32), *(v8h*)(kr + 48));
    v8f s = {};
    s = __builtin_amdgcn_wmma_f32_16x16x32_f16(false, a0, false, b0, (short)0, s, false, false);
    s = __builtin_amdgcn_wmma_f32_16x16x32_f16(false, a1, false, b1, (short)0, s, false, false);
#pragma unroll
    for (int j = 0; j < 8; ++j) {
      float v = s[j];
      Sbuf[(j + 8 * hi) * SROW + key0 + r] = (_Float16)v;
      m[j] = fmaxf(m[j], v);
    }
  }
  // max across the 16-lane half (rows live per lane-half), then publish
#pragma unroll
  for (int mask = 1; mask <= 8; mask <<= 1)
#pragma unroll
    for (int j = 0; j < 8; ++j) m[j] = fmaxf(m[j], __shfl_xor(m[j], mask, 32));
  if (r == 0) {
#pragma unroll
    for (int j = 0; j < 8; ++j)
      *(float*)(smem + (size_t)(j + 8 * hi) * SROW * 2 + 4096 + wid * 8) = m[j];
  }
  __syncthreads();
  // global max for this lane's P-row r
  float Mr = -1e30f;
#pragma unroll
  for (int w = 0; w < 4; ++w)
    Mr = fmaxf(Mr, *(const float*)(smem + (size_t)r * SROW * 2 + 4096 + w * 8));

  // ---- Pass 1.5: e = exp(s - Mr), store back to LDS, accumulate row sum ----
  const int kb = hi * 8;
  float lsum = 0.f;
  for (int st = 0; st < 16; ++st) {
    const int key0 = c0 + st * 32;
    _Float16* sp0 = &Sbuf[r * SROW + key0 + kb];
    _Float16* sp1 = &Sbuf[r * SROW + key0 + kb + 16];
    v8h s0 = *(v8h*)sp0, s1 = *(v8h*)sp1;
    v8h e0, e1;
#pragma unroll
    for (int i = 0; i < 8; ++i) {
      float e = __expf((float)s0[i] - Mr);
      lsum += e; e0[i] = (_Float16)e;
    }
#pragma unroll
    for (int i = 0; i < 8; ++i) {
      float e = __expf((float)s1[i] - Mr);
      lsum += e; e1[i] = (_Float16)e;
    }
    *(v8h*)sp0 = e0; *(v8h*)sp1 = e1;
  }
  lsum += __shfl_xor(lsum, 16, 32);            // lanes r and r+16 cover all cols
  if (hi == 0)
    *(float*)(smem + (size_t)r * SROW * 2 + 4096 + wid * 8 + 4) = lsum;
  __syncthreads();
  float Sr = 0.f;
#pragma unroll
  for (int w = 0; w < 4; ++w)
    Sr += *(const float*)(smem + (size_t)r * SROW * 2 + 4096 + w * 8 + 4);
  const float invS = 1.0f / Sr;
  const _Float16 invSh = (_Float16)invS;

  // ---- Pass 2: p = e*invS -> attn (NT) + O = P @ Vt (V pre-scaled 1/SEQL) ----
  float* attnRow = attn + (bh * SEQL + (size_t)qt * 16 + r) * SEQL;
  v8f o[4] = {};
  for (int st = 0; st < 16; ++st) {
    const int key0 = c0 + st * 32;
    v8h e0 = *(v8h*)&Sbuf[r * SROW + key0 + kb];
    v8h e1 = *(v8h*)&Sbuf[r * SROW + key0 + kb + 16];
    v16h pa;                                   // packed f16 multiplies
#pragma unroll
    for (int i = 0; i < 8; ++i) pa[i]     = e0[i] * invSh;
#pragma unroll
    for (int i = 0; i < 8; ++i) pa[8 + i] = e1[i] * invSh;
    // attn output (f32), write-once stream: keep it out of L2 with TH_NT
    v4f p0 = {(float)pa[0],  (float)pa[1],  (float)pa[2],  (float)pa[3]};
    v4f p1 = {(float)pa[4],  (float)pa[5],  (float)pa[6],  (float)pa[7]};
    v4f p2 = {(float)pa[8],  (float)pa[9],  (float)pa[10], (float)pa[11]};
    v4f p3 = {(float)pa[12], (float)pa[13], (float)pa[14], (float)pa[15]};
    __builtin_nontemporal_store(p0, (v4f*)(attnRow + key0 + kb));
    __builtin_nontemporal_store(p1, (v4f*)(attnRow + key0 + kb + 4));
    __builtin_nontemporal_store(p2, (v4f*)(attnRow + key0 + kb + 16));
    __builtin_nontemporal_store(p3, (v4f*)(attnRow + key0 + kb + 20));
    const int kbB = hi * 8;
#pragma unroll
    for (int nt = 0; nt < 4; ++nt) {
      const _Float16* vr = Vbase + (size_t)(nt * 16 + r) * SEQL + key0 + kbB;
      __builtin_prefetch((const void*)(vr + 32), 0, 0);
      v16h bb = pack16(*(v8h*)vr, *(v8h*)(vr + 16));
      o[nt] = __builtin_amdgcn_wmma_f32_16x16x32_f16(
          false, pa, false, bb, (short)0, o[nt], false, false);
    }
  }
  __syncthreads();               // Sbuf no longer needed: alias O accumulator
  float* Oa = (float*)smem;      // 16 x 64 f32
  for (int i = t; i < 16 * 64; i += 128) Oa[i] = 0.f;
  __syncthreads();
#pragma unroll
  for (int nt = 0; nt < 4; ++nt)
#pragma unroll
    for (int j = 0; j < 8; ++j)
      atomicAdd(&Oa[(j + 8 * hi) * 64 + nt * 16 + r], o[nt][j]);  // ds_add_f32
  __syncthreads();
  {
    int row = t >> 3, dd = (t & 7) * 8;
    float* dst = attn_v + ((size_t)b * SEQL + qt * 16 + row) * (NH * HD) + h * HD + dd;
    v4f o0 = *(v4f*)&Oa[row * 64 + dd];
    v4f o1 = *(v4f*)&Oa[row * 64 + dd + 4];
    __builtin_nontemporal_store(o0, (v4f*)dst);
    __builtin_nontemporal_store(o1, (v4f*)(dst + 4));
  }
}

// ---------------------------------------------------------------------------
extern "C" void kernel_launch(void* const* d_in, const int* in_sizes, int n_in,
                              void* d_out, int out_size, void* d_ws, size_t ws_size,
                              hipStream_t stream) {
  (void)in_sizes; (void)n_in; (void)out_size; (void)ws_size;
  const float* key   = (const float*)d_in[0];
  const float* query = (const float*)d_in[1];
  const float* value = (const float*)d_in[2];
  const float* Wk    = (const float*)d_in[3];
  const float* Wq    = (const float*)d_in[4];
  const float* Wv    = (const float*)d_in[5];

  _Float16* ws = (_Float16*)d_ws;                 // 24 MB of f16 scratch
  const size_t TEN = (size_t)NB * NH * SEQL * HD; // 4,194,304 halves each
  _Float16* Qh = ws;
  _Float16* Kh = ws + TEN;
  _Float16* Vt = ws + 2 * TEN;

  float* attn_v = (float*)d_out;
  float* attn   = attn_v + (size_t)NB * SEQL * NH * HD;

  proj_kernel<<<dim3(NH, (NB * SEQL) / 64, 3), 128, 0, stream>>>(
      query, key, value, Wq, Wk, Wv, Qh, Kh, Vt);

  attn_kernel<<<dim3(SEQL / 16, NH, NB), 128, 16 * 2064 * 2 /* 66048 B */, stream>>>(
      Qh, Kh, Vt, attn_v, attn);
}
